// FeedForwardEnsemble_6373731467988
// MI455X (gfx1250) — matrix operand order repaired and checked by reading, not compiled, not run on hardware
//
#include <hip/hip_runtime.h>
#include <hip/hip_bf16.h>

#define DIMN 1024
#define NUME 16
#define NB   4096

typedef __attribute__((ext_vector_type(16))) __bf16 v16bf;
typedef __attribute__((ext_vector_type(8)))  float  v8f;
typedef __attribute__((ext_vector_type(4)))  unsigned int u32x4;
typedef __attribute__((ext_vector_type(8)))  int i32x8;
typedef __attribute__((ext_vector_type(4)))  int i32x4;

#if defined(__has_builtin)
#  if __has_builtin(__builtin_amdgcn_tensor_load_to_lds) && __has_builtin(__builtin_amdgcn_s_wait_tensorcnt)
#    define HAVE_TDM 1
#  else
#    define HAVE_TDM 0
#  endif
#else
#  define HAVE_TDM 0
#endif

__device__ __forceinline__ unsigned short f2bf(float f) {
  union { float f; unsigned u; } c; c.f = f;
  unsigned u = c.u;
  u += 0x7FFFu + ((u >> 16) & 1u);   // round-to-nearest-even
  return (unsigned short)(u >> 16);
}

union FragBF {
  v16bf v;
  struct { uint4 lo; uint4 hi; } q;
};

#if HAVE_TDM
// Issue one TDM 2D tile load: global (row-major, stride0 elems of 2B) -> LDS
// contiguous [tile1][tile0].  D# packing per CDNA5 ISA.
__device__ __forceinline__ void tdm_load_2d(unsigned ldsOff, const void* gptr,
                                            unsigned tile0, unsigned tile1,
                                            unsigned long long stride0) {
  const unsigned long long ga = (unsigned long long)(size_t)gptr;
  const unsigned td0 = 0x40000000u, td1 = 0x40000000u;  // never clip
  u32x4 g0;
  g0[0] = 1u;                                          // count=1, no gather
  g0[1] = ldsOff;                                      // lds_addr (bytes)
  g0[2] = (unsigned)(ga & 0xFFFFFFFFull);              // global_addr[31:0]
  g0[3] = (unsigned)((ga >> 32) & 0x01FFFFFFull)       // global_addr[56:32]
        | (2u << 30);                                  // type=2 ("image")
  i32x8 g1;
  g1[0] = (int)(1u << 16);                             // data_size=1 (2 bytes)
  g1[1] = (int)((td0 & 0xFFFFu) << 16);                // tensor_dim0[15:0]
  g1[2] = (int)((td0 >> 16) | ((td1 & 0xFFFFu) << 16));// td0[31:16], td1[15:0]
  g1[3] = (int)((td1 >> 16) | (tile0 << 16));          // td1[31:16], tile_dim0
  g1[4] = (int)(tile1 & 0xFFFFu);                      // tile_dim1, tile_dim2=0
  g1[5] = (int)(unsigned)(stride0 & 0xFFFFFFFFull);    // dim0_stride[31:0]
  g1[6] = (int)(unsigned)((stride0 >> 32) & 0xFFFFull);// dim0_stride[47:32]
  g1[7] = 0;
  i32x4 z4 = {0, 0, 0, 0};
#if __clang_major__ >= 23
  i32x8 z8 = {0, 0, 0, 0, 0, 0, 0, 0};
  __builtin_amdgcn_tensor_load_to_lds(g0, g1, z4, z4, z8, 0);
#else
  __builtin_amdgcn_tensor_load_to_lds(g0, g1, z4, z4, 0);
#endif
}
#endif

// ---------------------------------------------------------------------------
// LayerNorm over last dim (1024) -> bf16.  One wave32 per row, 8 rows/block.
// ---------------------------------------------------------------------------
__global__ __launch_bounds__(256) void ln_to_bf16(const float* __restrict__ x,
                                                  const float* __restrict__ gamma,
                                                  const float* __restrict__ beta,
                                                  unsigned short* __restrict__ xnb) {
  const int lane = threadIdx.x & 31;
  const int wave = threadIdx.x >> 5;
  const size_t row = (size_t)blockIdx.x * 8 + wave;     // in [0, NB*NUME)
  const int e = (int)(row & (NUME - 1));                // ensemble member
  const float4* xr = (const float4*)(x + row * DIMN);
  const float4* gr = (const float4*)(gamma + (size_t)e * DIMN);
  const float4* br = (const float4*)(beta  + (size_t)e * DIMN);

  float4 v[8];
  float s = 0.f;
#pragma unroll
  for (int i = 0; i < 8; ++i) {
    v[i] = xr[lane + 32 * i];
    s += v[i].x + v[i].y + v[i].z + v[i].w;
  }
#pragma unroll
  for (int o = 16; o > 0; o >>= 1) s += __shfl_xor(s, o, 32);
  const float mu = s * (1.f / DIMN);

  float q = 0.f;
#pragma unroll
  for (int i = 0; i < 8; ++i) {
    float d0 = v[i].x - mu, d1 = v[i].y - mu, d2 = v[i].z - mu, d3 = v[i].w - mu;
    q += d0 * d0 + d1 * d1 + d2 * d2 + d3 * d3;
  }
#pragma unroll
  for (int o = 16; o > 0; o >>= 1) q += __shfl_xor(q, o, 32);
  const float inv = rsqrtf(q * (1.f / DIMN) + 1e-5f);

  unsigned short* orow = xnb + row * DIMN;
#pragma unroll
  for (int i = 0; i < 8; ++i) {
    const int j = lane + 32 * i;
    const float4 g4 = gr[j], b4 = br[j];
    ushort4 o;
    o.x = f2bf((v[i].x - mu) * inv * g4.x + b4.x);
    o.y = f2bf((v[i].y - mu) * inv * g4.y + b4.y);
    o.z = f2bf((v[i].z - mu) * inv * g4.z + b4.z);
    o.w = f2bf((v[i].w - mu) * inv * g4.w + b4.w);
    *(ushort4*)(orow + j * 4) = o;
  }
}

// ---------------------------------------------------------------------------
// W[n][d][k] f32  ->  WT[n][k][d] bf16   (so GEMM B-tiles are K-contiguous)
// ---------------------------------------------------------------------------
__global__ __launch_bounds__(256) void w_transpose_bf16(const float* __restrict__ W,
                                                        unsigned short* __restrict__ WT) {
  __shared__ unsigned short t[32][33];
  const int e  = blockIdx.z;
  const int k0 = blockIdx.x * 32;
  const int d0 = blockIdx.y * 32;
  const float*       We = W  + (size_t)e * DIMN * DIMN;
  unsigned short*    Te = WT + (size_t)e * DIMN * DIMN;
  const int tx = threadIdx.x & 31, ty = threadIdx.x >> 5;
#pragma unroll
  for (int r = 0; r < 32; r += 8)
    t[ty + r][tx] = f2bf(We[(size_t)(d0 + ty + r) * DIMN + k0 + tx]);
  __syncthreads();
#pragma unroll
  for (int r = 0; r < 32; r += 8)
    Te[(size_t)(k0 + ty + r) * DIMN + d0 + tx] = t[tx][ty + r];
}

// ---------------------------------------------------------------------------
// WMMA bf16 GEMM:  C[m,n'] = sum_k A[m,k] * WT[n',k]   (per ensemble member)
// Block: 128x128 tile, 8 waves, each wave 32(M) x 64(N), K-step 32.
// Tiles staged by the Tensor Data Mover (double-buffered LDS), wave 0 is the
// producer; explicit even/odd phases keep the WMMA accumulators in place.
// ---------------------------------------------------------------------------
template <bool FUSE_GELU>
__global__ __launch_bounds__(256) void gemm_wmma(const unsigned short* __restrict__ A,   // bf16 [NB][NUME][DIMN]
                                                 const unsigned short* __restrict__ Wt,  // bf16 [NUME][DIMN][DIMN] (K-major)
                                                 const float* __restrict__ bias,         // f32  [NUME][DIMN]
                                                 unsigned short* __restrict__ outBF,
                                                 float* __restrict__ outF) {
#if HAVE_TDM
  __shared__ unsigned short As[2][128 * 32];
  __shared__ unsigned short Bs[2][128 * 32];
#else
  __shared__ unsigned short As[1][128 * 32];
  __shared__ unsigned short Bs[1][128 * 32];
#endif

  const int e    = blockIdx.z;
  const int n0   = blockIdx.x * 128;
  const int m0   = blockIdx.y * 128;
  const int lane = threadIdx.x & 31;
  const int wave = threadIdx.x >> 5;
  const int waveM = (wave >> 1) * 32;
  const int waveN = (wave & 1) * 64;

  const unsigned short* We = Wt + (size_t)e * DIMN * DIMN;
  const float*          be = bias + (size_t)e * DIMN;
  const unsigned short* Ag = A + ((size_t)m0 * NUME + e) * DIMN;  // row m0, k=0
  const unsigned short* Bg = We + (size_t)n0 * DIMN;              // row n0, k=0

  v8f acc[2][4];
  const v8f vz = {0.f, 0.f, 0.f, 0.f, 0.f, 0.f, 0.f, 0.f};
#pragma unroll
  for (int i = 0; i < 2; ++i)
#pragma unroll
    for (int j = 0; j < 4; ++j) acc[i][j] = vz;

  const int l15 = lane & 15;
  const int hiA = (lane >> 4) * 8;    // A frag: lanes>=16 hold K 8..15 / 24..31
  const int hkB = (lane >> 4) * 8;    // B frag: lanes>=16 hold K 8..15 / 24..31

  const int NT = DIMN / 32;           // 32 K-steps (even)

  // consume one staged 128x32 tile pair: 6 frag loads, 8 WMMAs
  auto compute_tile = [&](const unsigned short* Ab, const unsigned short* Bb) {
    FragBF fa[2], fb[4];
#pragma unroll
    for (int i = 0; i < 2; ++i) {     // A 16x32 frag per ISA layout
      const unsigned short* pA = Ab + (waveM + i * 16 + l15) * 32 + hiA;
      fa[i].q.lo = *(const uint4*)pA;         // K {0..7} | {8..15}
      fa[i].q.hi = *(const uint4*)(pA + 16);  // K {16..23} | {24..31}
    }
#pragma unroll
    for (int j = 0; j < 4; ++j) {     // B 32x16 frag: same K order as A frag
      const unsigned short* pB = Bb + (waveN + j * 16 + l15) * 32 + hkB;
      fb[j].q.lo = *(const uint4*)pB;
      fb[j].q.hi = *(const uint4*)(pB + 16);
    }
#pragma unroll
    for (int i = 0; i < 2; ++i)
#pragma unroll
      for (int j = 0; j < 4; ++j)
        acc[i][j] = __builtin_amdgcn_wmma_f32_16x16x32_bf16(
            false, fa[i].v, false, fb[j].v, (short)0, acc[i][j], false, false);
  };

#if HAVE_TDM
  const bool wv0 = (wave == 0);
  const unsigned long long strideA = (unsigned long long)NUME * DIMN;  // elems
  if (wv0) {
    tdm_load_2d((unsigned)(size_t)&As[0][0], Ag, 32u, 128u, strideA);
    tdm_load_2d((unsigned)(size_t)&Bs[0][0], Bg, 32u, 128u, (unsigned long long)DIMN);
  }
  for (int kt = 0; kt < NT; kt += 2) {
    // ---- phase 0: stage kt+1 -> buf1, compute buf0 (kt) ----
    if (wv0) {
      tdm_load_2d((unsigned)(size_t)&As[1][0], Ag + (size_t)(kt + 1) * 32, 32u, 128u, strideA);
      tdm_load_2d((unsigned)(size_t)&Bs[1][0], Bg + (size_t)(kt + 1) * 32, 32u, 128u, (unsigned long long)DIMN);
      __builtin_amdgcn_s_wait_tensorcnt(2);  // pair(kt) done; pair(kt+1) in flight
    }
    __syncthreads();
    compute_tile(As[0], Bs[0]);
    __syncthreads();
    // ---- phase 1: stage kt+2 -> buf0, compute buf1 (kt+1) ----
    if (wv0) {
      if (kt + 2 < NT) {
        tdm_load_2d((unsigned)(size_t)&As[0][0], Ag + (size_t)(kt + 2) * 32, 32u, 128u, strideA);
        tdm_load_2d((unsigned)(size_t)&Bs[0][0], Bg + (size_t)(kt + 2) * 32, 32u, 128u, (unsigned long long)DIMN);
        __builtin_amdgcn_s_wait_tensorcnt(2);
      } else {
        __builtin_amdgcn_s_wait_tensorcnt(0);
      }
    }
    __syncthreads();
    compute_tile(As[1], Bs[1]);
    __syncthreads();
  }
#else
  for (int kt = 0; kt < NT; ++kt) {
    const int k0 = kt * 32;
    __syncthreads();
#pragma unroll
    for (int it = 0; it < 2; ++it) {
      const int c = threadIdx.x + it * 256;
      const int mr = c >> 2, seg = c & 3;
      const uint4* sa = (const uint4*)(A + ((size_t)(m0 + mr) * NUME + e) * DIMN + k0) + seg;
      ((uint4*)(As[0] + mr * 32))[seg] = *sa;
      const uint4* sb = (const uint4*)(We + (size_t)(n0 + mr) * DIMN + k0) + seg;
      ((uint4*)(Bs[0] + mr * 32))[seg] = *sb;
      if (k0 + 32 < DIMN) {
        __builtin_prefetch((const char*)sa + 64, 0, 3);
        __builtin_prefetch((const char*)sb + 64, 0, 3);
      }
    }
    __syncthreads();
    compute_tile(As[0], Bs[0]);
  }
#endif

  // epilogue: C/D layout -> lanes 0-15 rows M=r, lanes 16-31 rows M=r+8
  const int rowHi = (lane >> 4) * 8;
#pragma unroll
  for (int i = 0; i < 2; ++i) {
#pragma unroll
    for (int j = 0; j < 4; ++j) {
      const int colg = n0 + waveN + j * 16 + l15;
      const float bv = be[colg];
#pragma unroll
      for (int r = 0; r < 8; ++r) {
        const int rowg = m0 + waveM + i * 16 + rowHi + r;
        float val = acc[i][j][r] + bv;
        const size_t idx = ((size_t)rowg * NUME + e) * DIMN + colg;
        if (FUSE_GELU) {
          val = 0.5f * val * (1.f + erff(val * 0.70710678118654752f));
          outBF[idx] = f2bf(val);
        } else {
          outF[idx] = val;
        }
      }
    }
  }
}

// ---------------------------------------------------------------------------
extern "C" void kernel_launch(void* const* d_in, const int* in_sizes, int n_in,
                              void* d_out, int out_size, void* d_ws, size_t ws_size,
                              hipStream_t stream) {
  const float* x     = (const float*)d_in[0];
  const float* gamma = (const float*)d_in[1];
  const float* beta  = (const float*)d_in[2];
  const float* W1    = (const float*)d_in[3];
  const float* b1    = (const float*)d_in[4];
  const float* W2    = (const float*)d_in[5];
  const float* b2    = (const float*)d_in[6];
  float* out = (float*)d_out;

  const size_t actElems = (size_t)NB * NUME * DIMN;   // 64Mi
  const size_t wElems   = (size_t)NUME * DIMN * DIMN; // 16Mi
  unsigned short* xnb = (unsigned short*)d_ws;        // bf16 LayerNorm output
  unsigned short* hb  = xnb + actElems;               // bf16 hidden
  unsigned short* w1t = hb + actElems;                // bf16 W1^T
  unsigned short* w2t = w1t + wElems;                 // bf16 W2^T

  w_transpose_bf16<<<dim3(32, 32, 16), 256, 0, stream>>>(W1, w1t);
  w_transpose_bf16<<<dim3(32, 32, 16), 256, 0, stream>>>(W2, w2t);
  ln_to_bf16<<<dim3(NB * NUME / 8), 256, 0, stream>>>(x, gamma, beta, xnb);
  gemm_wmma<true ><<<dim3(8, 32, 16), 256, 0, stream>>>(xnb, w1t, b1, hb, nullptr);
  gemm_wmma<false><<<dim3(8, 32, 16), 256, 0, stream>>>(hb,  w2t, b2, nullptr, out);
}
